// Local_MessagePassing_s_77386720739480
// MI455X (gfx1250) — matrix operand order repaired and checked by reading, not compiled, not run on hardware
//
#include <hip/hip_runtime.h>
#include <hip/hip_bf16.h>
#include <cstdint>

typedef __bf16 bf16_t;
typedef __attribute__((ext_vector_type(16))) __bf16 v16bf;
typedef __attribute__((ext_vector_type(8)))  __bf16 v8bf;
typedef __attribute__((ext_vector_type(4)))  __bf16 v4bf;
typedef __attribute__((ext_vector_type(8)))  float  v8f;

// ---------------------------------------------------------------------------
// CDNA5 helpers
// ---------------------------------------------------------------------------
__device__ __forceinline__ unsigned lds_off_of(const void* p) {
  // low 32 bits of a flat LDS-aperture address are the LDS byte offset
  return (unsigned)(uintptr_t)p;
}
__device__ __forceinline__ void async_b128(unsigned lds_off, const void* gptr) {
  asm volatile("global_load_async_to_lds_b128 %0, %1, off"
               :: "v"(lds_off), "v"(gptr) : "memory");
}
__device__ __forceinline__ void wait_async0() {
  asm volatile("s_wait_asynccnt 0" ::: "memory");
}
__device__ __forceinline__ float silu_f(float v) {
  return v * (1.0f / (1.0f + __expf(-v)));
}

// ---------------------------------------------------------------------------
// Pack fp32 weights [K,128] into WMMA bf16 B-fragment order:
// slab(ks: 128 K) -> ktile(t: 32 K) -> ntile(n: 16 N) -> lane(32) -> 16 bf16
// Lane l holds column N = n*16 + (l&15); K = t*32 + (j>>3)*16 + (l>>4)*8 + (j&7), j=0..15
// ---------------------------------------------------------------------------
__global__ void k_pack(const float* __restrict__ W, bf16_t* __restrict__ dst, int K) {
  int e = blockIdx.x * 256 + threadIdx.x;
  if (e >= K * 128) return;
  int j  = e & 15;
  int l  = (e >> 4) & 31;
  int n  = (e >> 9) & 7;
  int t  = (e >> 12) & 3;
  int ks = e >> 14;
  int kk = ks * 128 + t * 32 + ((j >> 3) * 16) + ((l >> 4) * 8) + (j & 7);
  int nn = n * 16 + (l & 15);
  dst[e] = (bf16_t)W[(size_t)kk * 128 + nn];
}

// ---------------------------------------------------------------------------
// fp32 -> bf16 bulk convert (8 elements / thread)
// ---------------------------------------------------------------------------
__global__ void k_cvt(const float* __restrict__ src, bf16_t* __restrict__ dst, long n8) {
  long t = (long)blockIdx.x * 256 + threadIdx.x;
  if (t >= n8) return;
  const float4* s = (const float4*)src + t * 2;
  float4 f0 = s[0], f1 = s[1];
  v8bf r;
  r[0] = (bf16_t)f0.x; r[1] = (bf16_t)f0.y; r[2] = (bf16_t)f0.z; r[3] = (bf16_t)f0.w;
  r[4] = (bf16_t)f1.x; r[5] = (bf16_t)f1.y; r[6] = (bf16_t)f1.z; r[7] = (bf16_t)f1.w;
  ((v8bf*)dst)[t] = r;
}

// ---------------------------------------------------------------------------
// Generic 128-col WMMA GEMM: out = [silu](A[M,K] x Wpacked[K,128] + bias) (+res0)(+res1)
// block = 256 threads (8 wave32), tile 128 rows x 128 cols, K in 128-slabs.
// A slab + W slab staged in LDS; W (and bf16-A) staged via async-to-LDS DMA.
// Inner loop is software-pipelined: B fragment s+1 (and A fragment t+1) are
// loaded before the WMMA consuming fragment s, forcing distinct registers so
// one LDS load stays in flight under every WMMA (s_wait_dscnt <= 1).
// ---------------------------------------------------------------------------
__global__ __launch_bounds__(256)
void k_gemm(const void* __restrict__ Asrc, int a_is_f32, int K, int M,
            const bf16_t* __restrict__ Wp, const float* __restrict__ bias,
            int do_silu,
            const float* __restrict__ res0, const float* __restrict__ res1,
            float* __restrict__ outF, bf16_t* __restrict__ outB)
{
  __shared__ bf16_t sA[128 * 128];   // 32 KB: A slab, row-major
  __shared__ bf16_t sW[128 * 128];   // 32 KB: packed W slab (fragment order)

  const int tid   = threadIdx.x;
  const int lane  = tid & 31;
  const int wv    = tid >> 5;
  const int mbase = blockIdx.x * 128;

  v8f acc[8];
  #pragma unroll
  for (int n = 0; n < 8; ++n) {
    #pragma unroll
    for (int r = 0; r < 8; ++r) acc[n][r] = 0.0f;
  }

  const int nslab = K >> 7;

  const int arowbase = wv * 16 + (lane & 15);
  const int kb       = (lane >> 4) << 3;           // 0 or 8

  for (int ks = 0; ks < nslab; ++ks) {
    // ---- stage packed W slab into LDS via async DMA (128 B / thread) ----
    {
      const uint4* gw = (const uint4*)(Wp + (size_t)ks * 16384);
      #pragma unroll
      for (int v = 0; v < 8; ++v) {
        int chunk = tid * 8 + v;
        async_b128(lds_off_of(&sW[chunk * 8]), gw + chunk);
      }
    }
    if (ks + 1 < nslab) __builtin_prefetch(Wp + (size_t)(ks + 1) * 16384, 0, 0);

    // ---- stage A slab: 128 rows x 128 K (2 threads / row) ----
    {
      int row  = tid >> 1;
      int half = tid & 1;
      long rowg = (long)mbase + row;
      if (rowg >= M) rowg = M - 1;                 // clamp tail reads
      if (a_is_f32) {
        const float* src = (const float*)Asrc + (size_t)rowg * K + ks * 128 + half * 64;
        #pragma unroll
        for (int v = 0; v < 16; ++v) {
          float4 f = ((const float4*)src)[v];
          v4bf t4;
          t4[0] = (bf16_t)f.x; t4[1] = (bf16_t)f.y;
          t4[2] = (bf16_t)f.z; t4[3] = (bf16_t)f.w;
          *(v4bf*)&sA[row * 128 + half * 64 + v * 4] = t4;
        }
      } else {
        const uint4* src =
            (const uint4*)((const bf16_t*)Asrc + (size_t)rowg * K + ks * 128 + half * 64);
        #pragma unroll
        for (int v = 0; v < 8; ++v)
          async_b128(lds_off_of(&sA[row * 128 + half * 64 + v * 8]), src + v);
      }
    }
    wait_async0();        // this wave's async-to-LDS done
    __syncthreads();      // all waves' slabs visible

    // ---- software-pipelined WMMA sweep: 32 steps = 4 ktiles x 8 ntiles ----
    auto loadA = [&](int t) -> v16bf {
      const bf16_t* ap = &sA[arowbase * 128 + t * 32 + kb];
      v8bf lo = *(const v8bf*)ap;                  // K = kb .. kb+7
      v8bf hi = *(const v8bf*)(ap + 16);           // K = kb+16 .. kb+23
      v16bf a;
      #pragma unroll
      for (int q = 0; q < 8; ++q) { a[q] = lo[q]; a[8 + q] = hi[q]; }
      return a;
    };
    auto loadB = [&](int s) -> v16bf {
      return *(const v16bf*)&sW[(s * 32 + lane) * 16];
    };

    v16bf a    = loadA(0);
    v16bf bcur = loadB(0);
    #pragma unroll
    for (int s = 0; s < 32; ++s) {
      const int n = s & 7;
      v16bf bnext = bcur;
      v16bf anext = a;
      if (s < 31)                  bnext = loadB(s + 1);
      if ((s & 7) == 7 && s < 31)  anext = loadA((s >> 3) + 1);
      acc[n] = __builtin_amdgcn_wmma_f32_16x16x32_bf16(
          false, a, false, bcur, (short)0, acc[n], false, false);
      bcur = bnext;
      a    = anext;
    }
    __syncthreads();
  }

  // ---- epilogue: bias + SiLU + residuals, dual f32 / bf16 stores ----
  const int colb = lane & 15;
  const int rhi  = (lane >> 4) << 3;               // +8 rows for upper half-wave
  #pragma unroll
  for (int n = 0; n < 8; ++n) {
    const int col = n * 16 + colb;
    const float bcol = bias ? bias[col] : 0.0f;
    #pragma unroll
    for (int r = 0; r < 8; ++r) {
      const long row = (long)mbase + wv * 16 + rhi + r;
      if (row < M) {
        float v = acc[n][r] + bcol;
        if (do_silu) v = silu_f(v);
        size_t off = (size_t)row * 128 + col;
        if (res0) v += res0[off];
        if (res1) v += res1[off];
        if (outF) outF[off] = v;
        if (outB) outB[off] = (bf16_t)v;
      }
    }
  }
}

// ---------------------------------------------------------------------------
// m = concat(x1[i], x1[j], rbf) as bf16 rows of 384 (16B chunk per thread)
// ---------------------------------------------------------------------------
__global__ void k_concat(const bf16_t* __restrict__ xB, const bf16_t* __restrict__ rbfB,
                         const int* __restrict__ eidx, long E, bf16_t* __restrict__ mB) {
  long t = (long)blockIdx.x * 256 + threadIdx.x;
  if (t >= E * 48) return;
  long e = t / 48; int c = (int)(t % 48);
  const uint4* src;
  if (c < 16)      src = (const uint4*)(xB + (size_t)eidx[E + e] * 128) + c;        // x1[i]
  else if (c < 32) src = (const uint4*)(xB + (size_t)eidx[e] * 128) + (c - 16);     // x1[j]
  else             src = (const uint4*)(rbfB + (size_t)e * 128) + (c - 32);         // rbf
  ((uint4*)(mB + (size_t)e * 384))[c] = *src;
}

// elementwise bf16 product (8 / thread): m_neighbor = mjj * (rbf@lin_rbf)
__global__ void k_mul(const bf16_t* __restrict__ a, const bf16_t* __restrict__ b,
                      bf16_t* __restrict__ o, long n8) {
  long t = (long)blockIdx.x * 256 + threadIdx.x;
  if (t >= n8) return;
  v8bf x = ((const v8bf*)a)[t], y = ((const v8bf*)b)[t], r;
  #pragma unroll
  for (int k = 0; k < 8; ++k) r[k] = (bf16_t)((float)x[k] * (float)y[k]);
  ((v8bf*)o)[t] = r;
}

// pair stage: acc[idx_ji[p]] += m_neighbor[idx_jj[p]] * sbf2[p]   (P -> E segment sum)
__global__ void k_pair(const bf16_t* __restrict__ mnb, const bf16_t* __restrict__ sbf2,
                       const int* __restrict__ jj, const int* __restrict__ ji,
                       float* __restrict__ acc) {
  long p = blockIdx.x; int c = threadIdx.x;
  float v = (float)mnb[(size_t)jj[p] * 128 + c] * (float)sbf2[(size_t)p * 128 + c];
  atomicAdd(&acc[(size_t)ji[p] * 128 + c], v);
}

// edge stage: xacc[i[e]] += (rbf@lin_rbf_out)[e] * m[e]            (E -> N segment sum)
__global__ void k_escale(const bf16_t* __restrict__ rl, const float* __restrict__ m,
                         const int* __restrict__ eidx, long E, float* __restrict__ xacc) {
  long e = blockIdx.x; int c = threadIdx.x;
  float v = (float)rl[(size_t)e * 128 + c] * m[(size_t)e * 128 + c];
  atomicAdd(&xacc[(size_t)eidx[E + e] * 128 + c], v);
}

// final projections: out@W (att) and out@W_out + b (proj); one wave32 per row
__global__ void k_dots(const bf16_t* __restrict__ outB,
                       const float* __restrict__ Wv, const float* __restrict__ Wo,
                       const float* __restrict__ bo, long N,
                       float* __restrict__ dproj, float* __restrict__ datt) {
  long row = (long)blockIdx.x * 8 + (threadIdx.x >> 5);
  int lane = threadIdx.x & 31;
  if (row >= N) return;
  float s1 = 0.f, s2 = 0.f;
  #pragma unroll
  for (int k = 0; k < 4; ++k) {
    int c = lane + 32 * k;
    float o = (float)outB[(size_t)row * 128 + c];
    s1 += o * Wv[c];
    s2 += o * Wo[c];
  }
  #pragma unroll
  for (int off = 16; off > 0; off >>= 1) {
    s1 += __shfl_down(s1, off, 32);
    s2 += __shfl_down(s2, off, 32);
  }
  if (lane == 0) { dproj[row] = s2 + bo[0]; datt[row] = s1; }
}

// ---------------------------------------------------------------------------
extern "C" void kernel_launch(void* const* d_in, const int* in_sizes, int n_in,
                              void* d_out, int out_size, void* d_ws, size_t ws_size,
                              hipStream_t stream) {
  (void)n_in; (void)out_size; (void)ws_size;
  const long N = in_sizes[0] / 128;
  const long E = in_sizes[1] / 128;
  const long P = in_sizes[2] / 128;

  const float* x_in = (const float*)d_in[0];
  const float* rbf  = (const float*)d_in[1];
  const float* sbf  = (const float*)d_in[2];
  // params in setup_inputs() insertion order
  const float* W_x1 = (const float*)d_in[3],  *b_x1 = (const float*)d_in[4];
  const float* W_ji = (const float*)d_in[5],  *b_ji = (const float*)d_in[6];
  const float* W_jj = (const float*)d_in[7],  *b_jj = (const float*)d_in[8];
  const float* W_s1 = (const float*)d_in[9],  *b_s1 = (const float*)d_in[10];
  const float* W_s2 = (const float*)d_in[11], *b_s2 = (const float*)d_in[12];
  const float* W_rbf = (const float*)d_in[13];
  const float* W_r1a = (const float*)d_in[14], *b_r1a = (const float*)d_in[15];
  const float* W_r1b = (const float*)d_in[16], *b_r1b = (const float*)d_in[17];
  const float* W_r2a = (const float*)d_in[18], *b_r2a = (const float*)d_in[19];
  const float* W_r2b = (const float*)d_in[20], *b_r2b = (const float*)d_in[21];
  const float* W_r3a = (const float*)d_in[22], *b_r3a = (const float*)d_in[23];
  const float* W_r3b = (const float*)d_in[24], *b_r3b = (const float*)d_in[25];
  const float* W_rbo = (const float*)d_in[26];
  const float* W_x2  = (const float*)d_in[27], *b_x2 = (const float*)d_in[28];
  const float* W_o1  = (const float*)d_in[29], *b_o1 = (const float*)d_in[30];
  const float* W_o2  = (const float*)d_in[31], *b_o2 = (const float*)d_in[32];
  const float* W_o3  = (const float*)d_in[33], *b_o3 = (const float*)d_in[34];
  const float* WoutW = (const float*)d_in[35], *Woutb = (const float*)d_in[36];
  const float* W_att = (const float*)d_in[37];
  const int* idx_jj = (const int*)d_in[38];
  const int* idx_ji = (const int*)d_in[39];
  const int* eidx   = (const int*)d_in[40];

  // workspace bump allocator (256B aligned)
  char* w = (char*)d_ws; size_t off = 0;
  auto alloc = [&](size_t bytes) -> char* {
    char* p = w + off; off = (off + bytes + 255) & ~(size_t)255; return p;
  };
  bf16_t* pW_x1  = (bf16_t*)alloc(128 * 128 * 2);
  bf16_t* pW_ji  = (bf16_t*)alloc(384 * 128 * 2);
  bf16_t* pW_jj  = (bf16_t*)alloc(384 * 128 * 2);
  bf16_t* pW_s1  = (bf16_t*)alloc(128 * 128 * 2);
  bf16_t* pW_s2  = (bf16_t*)alloc(128 * 128 * 2);
  bf16_t* pW_rbf = (bf16_t*)alloc(128 * 128 * 2);
  bf16_t* pW_rbo = (bf16_t*)alloc(128 * 128 * 2);
  bf16_t* pW_r1a = (bf16_t*)alloc(128 * 128 * 2);
  bf16_t* pW_r1b = (bf16_t*)alloc(128 * 128 * 2);
  bf16_t* pW_r2a = (bf16_t*)alloc(128 * 128 * 2);
  bf16_t* pW_r2b = (bf16_t*)alloc(128 * 128 * 2);
  bf16_t* pW_r3a = (bf16_t*)alloc(128 * 128 * 2);
  bf16_t* pW_r3b = (bf16_t*)alloc(128 * 128 * 2);
  bf16_t* pW_x2  = (bf16_t*)alloc(128 * 128 * 2);
  bf16_t* pW_o1  = (bf16_t*)alloc(128 * 128 * 2);
  bf16_t* pW_o2  = (bf16_t*)alloc(128 * 128 * 2);
  bf16_t* pW_o3  = (bf16_t*)alloc(128 * 128 * 2);

  float*  x1F  = (float*) alloc((size_t)N * 128 * 4);
  bf16_t* x1B  = (bf16_t*)alloc((size_t)N * 128 * 2);
  float*  nFa  = (float*) alloc((size_t)N * 128 * 4);
  float*  nFb  = (float*) alloc((size_t)N * 128 * 4);
  bf16_t* nBa  = (bf16_t*)alloc((size_t)N * 128 * 2);
  bf16_t* nBb  = (bf16_t*)alloc((size_t)N * 128 * 2);
  bf16_t* rbfB = (bf16_t*)alloc((size_t)E * 128 * 2);
  bf16_t* rlB  = (bf16_t*)alloc((size_t)E * 128 * 2);
  bf16_t* qB   = (bf16_t*)alloc((size_t)E * 128 * 2);
  bf16_t* mnbB = (bf16_t*)alloc((size_t)E * 128 * 2);
  bf16_t* mB   = (bf16_t*)alloc((size_t)E * 384 * 2);
  float*  mjiF = (float*) alloc((size_t)E * 128 * 4);
  bf16_t* sbfB = (bf16_t*)alloc((size_t)P * 128 * 2);

  float* out_x    = (float*)d_out;            // [N,128]
  float* out_proj = out_x + (size_t)N * 128;  // [N]
  float* out_att  = out_proj + N;             // [N]

  auto pack = [&](const float* Ws, bf16_t* dst, int K) {
    k_pack<<<dim3((unsigned)((K * 128 + 255) / 256)), dim3(256), 0, stream>>>(Ws, dst, K);
  };
  auto gemm = [&](const void* A, int af32, int K, long M, const bf16_t* Wp,
                  const float* bias, int s, const float* r0, const float* r1,
                  float* oF, bf16_t* oB) {
    k_gemm<<<dim3((unsigned)((M + 127) / 128)), dim3(256), 0, stream>>>(
        A, af32, K, (int)M, Wp, bias, s, r0, r1, oF, oB);
  };
  auto cvt = [&](const float* s, bf16_t* d, long count) {
    long n8 = count / 8;
    k_cvt<<<dim3((unsigned)((n8 + 255) / 256)), dim3(256), 0, stream>>>(s, d, n8);
  };

  // ---- weight packing ----
  pack(W_x1, pW_x1, 128);  pack(W_ji, pW_ji, 384);  pack(W_jj, pW_jj, 384);
  pack(W_s1, pW_s1, 128);  pack(W_s2, pW_s2, 128);
  pack(W_rbf, pW_rbf, 128); pack(W_rbo, pW_rbo, 128);
  pack(W_r1a, pW_r1a, 128); pack(W_r1b, pW_r1b, 128);
  pack(W_r2a, pW_r2a, 128); pack(W_r2b, pW_r2b, 128);
  pack(W_r3a, pW_r3a, 128); pack(W_r3b, pW_r3b, 128);
  pack(W_x2, pW_x2, 128);
  pack(W_o1, pW_o1, 128);  pack(W_o2, pW_o2, 128);  pack(W_o3, pW_o3, 128);

  // ---- pipeline ----
  gemm(x_in, 1, 128, N, pW_x1, b_x1, 1, nullptr, nullptr, x1F, x1B);   // x1 = silu(x@W1+b)
  cvt(rbf, rbfB, E * 128);
  cvt(sbf, sbfB, P * 128);
  { long tt = E * 48;
    k_concat<<<dim3((unsigned)((tt + 255) / 256)), dim3(256), 0, stream>>>(
        x1B, rbfB, eidx, E, mB); }

  gemm(mB, 0, 384, E, pW_ji, b_ji, 1, nullptr, nullptr, mjiF, nullptr); // m_ji
  gemm(mB, 0, 384, E, pW_jj, b_jj, 1, nullptr, nullptr, nullptr, qB);   // silu(m@Wjj+b)
  gemm(rbfB, 0, 128, E, pW_rbf, nullptr, 0, nullptr, nullptr, nullptr, rlB); // rbf@lin_rbf
  { long n8 = E * 16;
    k_mul<<<dim3((unsigned)((n8 + 255) / 256)), dim3(256), 0, stream>>>(qB, rlB, mnbB, n8); }

  gemm(sbfB, 0, 128, P, pW_s1, b_s1, 1, nullptr, nullptr, nullptr, sbfB);   // sbf mlp L1 (in-place)
  gemm(sbfB, 0, 128, P, pW_s2, b_s2, 1, nullptr, nullptr, nullptr, sbfB);   // sbf mlp L2 (in-place)

  k_pair<<<dim3((unsigned)P), dim3(128), 0, stream>>>(mnbB, sbfB, idx_jj, idx_ji, mjiF);

  gemm(rbfB, 0, 128, E, pW_rbo, nullptr, 0, nullptr, nullptr, nullptr, rlB); // rbf@lin_rbf_out
  k_escale<<<dim3((unsigned)E), dim3(128), 0, stream>>>(rlB, mjiF, eidx, E, x1F);

  gemm(x1F, 1, 128, N, pW_x2, b_x2, 1, nullptr, nullptr, nFa, nBa);          // x2
  gemm(nBa, 0, 128, N, pW_r1a, b_r1a, 1, nullptr, nullptr, nullptr, nBb);
  gemm(nBb, 0, 128, N, pW_r1b, b_r1b, 1, nFa, x_in, nFb, nBa);               // x3 = x2+h+res_x
  gemm(nBa, 0, 128, N, pW_r2a, b_r2a, 1, nullptr, nullptr, nullptr, nBb);
  gemm(nBb, 0, 128, N, pW_r2b, b_r2b, 1, nFb, nullptr, nFa, nBa);            // x4
  gemm(nBa, 0, 128, N, pW_r3a, b_r3a, 1, nullptr, nullptr, nullptr, nBb);
  gemm(nBb, 0, 128, N, pW_r3b, b_r3b, 1, nFa, nullptr, out_x, nBa);          // x5 -> d_out
  gemm(nBa, 0, 128, N, pW_o1, b_o1, 1, nullptr, nullptr, nullptr, nBb);
  gemm(nBb, 0, 128, N, pW_o2, b_o2, 1, nullptr, nullptr, nullptr, nBa);
  gemm(nBa, 0, 128, N, pW_o3, b_o3, 1, nullptr, nullptr, nullptr, nBb);      // out

  k_dots<<<dim3((unsigned)((N + 7) / 8)), dim3(256), 0, stream>>>(
      nBb, W_att, WoutW, Woutb, N, out_proj, out_att);
}
